// MultiHeadLocalAttention_36086315221022
// MI455X (gfx1250) — matrix operand order repaired and checked by reading, not compiled, not run on hardware
//
#include <hip/hip_runtime.h>

typedef unsigned short ushort_t;
typedef __attribute__((ext_vector_type(16))) __bf16 v16bf;
typedef __attribute__((ext_vector_type(8)))  float  v8f;

union BFrag {
    v16bf    v;
    ushort_t u[16];
    uint4    q[2];
};

__device__ __forceinline__ ushort_t f2bf(float f) {
    unsigned u = __float_as_uint(f);
    unsigned r = (u + 0x7FFFu + ((u >> 16) & 1u)) >> 16;  // round-to-nearest-even
    return (ushort_t)r;
}

__device__ __forceinline__ v8f vzero8() {
    v8f z;
#pragma unroll
    for (int i = 0; i < 8; ++i) z[i] = 0.0f;
    return z;
}

__device__ __forceinline__ v8f wmma_bf16(const BFrag& a, const BFrag& b, v8f c) {
    return __builtin_amdgcn_wmma_f32_16x16x32_bf16(
        /*neg_a=*/false, a.v, /*neg_b=*/false, b.v,
        /*c_mod=*/(short)0, c, /*reuse_a=*/false, /*reuse_b=*/false);
}

// A-matrix 16x32 bf16 fragment from a row-major tile.
// rp = row*ld + koff. Pattern: e<8 -> k=half*8+e ; e>=8 -> k=16+half*8+(e-8)
__device__ __forceinline__ void load_a_row(BFrag& f, const ushort_t* rp, int half) {
    f.q[0] = *(const uint4*)(rp + half * 8);
    f.q[1] = *(const uint4*)(rp + 16 + half * 8);
}

// B-matrix 32x16 bf16 fragment from B^T rows (row-major): e<8 -> k=half*8+e ; e>=8 -> k=16+half*8+(e-8).
__device__ __forceinline__ void load_b_row(BFrag& f, const ushort_t* rp, int half) {
    f.q[0] = *(const uint4*)(rp + half * 8);
    f.q[1] = *(const uint4*)(rp + 16 + half * 8);
}

#define NB      4
#define NC      256
#define NH      224
#define NWID    224
#define HEADS   8
#define HD      32
#define GH      28
#define NWIN    784          // 28*28
#define NPIX    50176        // 224*224
#define TOKS    64
#define LDO     264          // padded LDS row stride (bf16 elems): 264*2=528B, 16B-aligned

// LDS map (halfword offsets):
//  region0 [0 .. 64*LDO)           : xw[tok][c] in P0/P1, then O[tok][c] in P4/P5
//  region1 [64*LDO .. +48*1024)    : per-head (head-major): h*6144 -> {qT | kT | vT} 2048 each,
//                                    qT+kT (4096) later overlaid by S (64x64 bf16)
#define R1OFF   (64 * LDO)
#define LDS_HALFWORDS (R1OFF + 48 * 1024)

// ---------------------------------------------------------------- weights f32 -> bf16
__global__ void k_cvt(const float* __restrict__ wq, const float* __restrict__ wp,
                      ushort_t* __restrict__ wqb, ushort_t* __restrict__ wpb) {
    int i = blockIdx.x * 256 + threadIdx.x;
    if (i < 768 * 256) wqb[i] = f2bf(wq[i]);
    if (i < 256 * 256) wpb[i] = f2bf(wp[i]);
}

// ------------------- fully fused: window -> QKV GEMM -> attention -> projection
__global__ __launch_bounds__(256) void k_fused(const float* __restrict__ x,
                                               const ushort_t* __restrict__ wqb,
                                               const ushort_t* __restrict__ wpb,
                                               const float* __restrict__ bias,
                                               float* __restrict__ out) {
    extern __shared__ __align__(16) ushort_t lds[];
    ushort_t* xwl = lds;            // region0
    ushort_t* qkv = lds + R1OFF;    // region1

    int tid = threadIdx.x;
    int wave = tid >> 5;
    int lane = tid & 31;
    int lane15 = lane & 15, half = lane >> 4;
    int win = blockIdx.x % NWIN, b = blockIdx.x / NWIN;
    int gh = win / GH, gw = win % GH;

    // ---- P0: stage x window -> LDS xw[tok][c] bf16 (coalesced global reads)
    const float* xb = x + (size_t)b * NC * NPIX + (size_t)(gh * 8) * NWID + gw * 8;
    for (int idx = tid; idx < NC * TOKS; idx += 256) {
        int ch = idx >> 6, tok = idx & 63;
        int wy = tok >> 3, wx = tok & 7;
        float v = xb[(size_t)ch * NPIX + wy * NWID + wx];
        xwl[tok * LDO + ch] = f2bf(v);
    }
    __syncthreads();

    // ---- P1/P2: QKV GEMM [768x256]@[256x64]; B from LDS, A from global (L2-hot).
    //      Results written to per-head q/k/vT LDS regions (transposed [d][tok]).
    for (int ct = 0; ct < 4; ++ct) {
        v8f acc[6];
#pragma unroll
        for (int r = 0; r < 6; ++r) acc[r] = vzero8();
        for (int k = 0; k < 8; ++k) {
            BFrag Bf;
            load_b_row(Bf, xwl + (ct * 16 + lane15) * LDO + k * 32, half);
#pragma unroll
            for (int r = 0; r < 6; ++r) {
                BFrag Af;
                load_a_row(Af, wqb + (size_t)((wave * 6 + r) * 16 + lane15) * NC + k * 32, half);
                acc[r] = wmma_bf16(Af, Bf, acc[r]);
            }
        }
        int tok = ct * 16 + lane15;
#pragma unroll
        for (int r = 0; r < 6; ++r) {
            int o0 = (wave * 6 + r) * 16;     // constant per tile
            int sel = o0 >> 8;                // 0=q 1=k 2=v
            int c0 = o0 & 255;
            int h = c0 >> 5, dbase = c0 & 31;
            ushort_t* dst = qkv + (h * 3 + sel) * 2048;   // head-major {q,k,vT}
#pragma unroll
            for (int rr = 0; rr < 8; ++rr) {
                int d = dbase + rr + 8 * half;
                dst[d * TOKS + tok] = f2bf(acc[r][rr]);
            }
        }
    }
    __syncthreads();

    // ---- P3: windowed attention, wave == head
    ushort_t* qT = qkv + wave * 6144;
    ushort_t* kT = qT + 2048;
    ushort_t* vT = kT + 2048;
    ushort_t* S  = qT;               // 64x64 bf16 overlay of qT+kT after frags built

    BFrag Aq[4], Bk[4];
#pragma unroll
    for (int j = 0; j < 4; ++j) {
        int t = j * 16 + lane15;
#pragma unroll
        for (int e = 0; e < 16; ++e) {
            int ka  = ((e & 8) ? 16 : 0) + half * 8 + (e & 7);  // A pattern (k = head dim)
            int kbp = ((e & 8) ? 16 : 0) + half * 8 + (e & 7);  // B pattern
            Aq[j].u[e] = qT[ka * TOKS + t];
            Bk[j].u[e] = kT[kbp * TOKS + t];
        }
    }
    BFrag Bv[4];                     // [n*2+kk]: B(k=tok, n=d) = vT[d][tok], contiguous
#pragma unroll
    for (int n = 0; n < 2; ++n)
#pragma unroll
        for (int kk = 0; kk < 2; ++kk)
            load_b_row(Bv[n * 2 + kk], vT + (n * 16 + lane15) * TOKS + kk * 32, half);

    const float scale = 0.17677669529663687f;   // 1/sqrt(32)

    for (int i = 0; i < 4; ++i) {
        v8f sj[4];
#pragma unroll
        for (int j = 0; j < 4; ++j) sj[j] = wmma_bf16(Aq[i], Bk[j], vzero8());
#pragma unroll
        for (int r = 0; r < 8; ++r) {
            float vv[4];
#pragma unroll
            for (int j = 0; j < 4; ++j) vv[j] = sj[j][r] * scale;
            float m = fmaxf(fmaxf(vv[0], vv[1]), fmaxf(vv[2], vv[3]));
#pragma unroll
            for (int mk = 8; mk >= 1; mk >>= 1) m = fmaxf(m, __shfl_xor(m, mk, 16));
            float e[4], s = 0.0f;
#pragma unroll
            for (int j = 0; j < 4; ++j) { e[j] = __expf(vv[j] - m); s += e[j]; }
#pragma unroll
            for (int mk = 8; mk >= 1; mk >>= 1) s += __shfl_xor(s, mk, 16);
            float inv = 1.0f / s;
            int row = i * 16 + r + 8 * half;
#pragma unroll
            for (int j = 0; j < 4; ++j) S[row * TOKS + j * 16 + lane15] = f2bf(e[j] * inv);
        }
    }

    // ---- P4: O = P @ V -> region0 O[tok][256] (xw is dead)
    for (int i = 0; i < 4; ++i) {
        BFrag Ap0, Ap1;
        const ushort_t* rp = S + (i * 16 + lane15) * TOKS;
        load_a_row(Ap0, rp, half);
        load_a_row(Ap1, rp + 32, half);
#pragma unroll
        for (int n = 0; n < 2; ++n) {
            v8f acc = vzero8();
            acc = wmma_bf16(Ap0, Bv[n * 2 + 0], acc);
            acc = wmma_bf16(Ap1, Bv[n * 2 + 1], acc);
#pragma unroll
            for (int rr = 0; rr < 8; ++rr) {
                int tok = i * 16 + rr + 8 * half;
                xwl[tok * LDO + wave * HD + n * 16 + lane15] = f2bf(acc[rr]);
            }
        }
    }
    __syncthreads();

    // ---- P5: projection GEMM [256x256]@[256x64] + bias -> NCHW f32
    for (int ct = 0; ct < 4; ++ct) {
        v8f acc[2];
        acc[0] = vzero8(); acc[1] = vzero8();
        for (int k = 0; k < 8; ++k) {
            BFrag Bf;
            load_b_row(Bf, xwl + (ct * 16 + lane15) * LDO + k * 32, half);
#pragma unroll
            for (int r = 0; r < 2; ++r) {
                BFrag Af;
                load_a_row(Af, wpb + (size_t)((wave * 2 + r) * 16 + lane15) * NC + k * 32, half);
                acc[r] = wmma_bf16(Af, Bf, acc[r]);
            }
        }
        int tok = ct * 16 + lane15;
        int wy = tok >> 3, wx = tok & 7;
        int pix = (gh * 8 + wy) * NWID + gw * 8 + wx;
#pragma unroll
        for (int r = 0; r < 2; ++r)
#pragma unroll
            for (int rr = 0; rr < 8; ++rr) {
                int o = (wave * 2 + r) * 16 + rr + 8 * half;
                out[((size_t)(b * NC + o)) * NPIX + pix] = acc[r][rr] + bias[o];
            }
    }
}

extern "C" void kernel_launch(void* const* d_in, const int* in_sizes, int n_in,
                              void* d_out, int out_size, void* d_ws, size_t ws_size,
                              hipStream_t stream) {
    (void)in_sizes; (void)n_in; (void)out_size; (void)ws_size;
    const float* x     = (const float*)d_in[0];
    const float* wqkv  = (const float*)d_in[1];
    const float* wproj = (const float*)d_in[2];
    const float* bias  = (const float*)d_in[3];
    float* out = (float*)d_out;

    char* ws = (char*)d_ws;
    ushort_t* wqb = (ushort_t*)ws;              // 768*256 bf16
    ushort_t* wpb = (ushort_t*)(ws + 393216);   // 256*256 bf16

    const size_t ldsBytes = (size_t)LDS_HALFWORDS * sizeof(ushort_t);   // 132,096 B
    // Request >64KB dynamic LDS (CDNA5 WGP supports 320KB). Host-side, capture-safe.
    static int attrDone = 0;
    (void)attrDone;
    hipFuncSetAttribute((const void*)k_fused,
                        hipFuncAttributeMaxDynamicSharedMemorySize, (int)ldsBytes);

    k_cvt  <<<768,       256, 0,        stream>>>(wqkv, wproj, wqb, wpb);
    k_fused<<<NB * NWIN, 256, ldsBytes, stream>>>(x, wqb, wpb, bias, out);
}
